// Pairwise_69698729279795
// MI455X (gfx1250) — matrix operand order repaired and checked by reading, not compiled, not run on hardware
//
#include <hip/hip_runtime.h>

typedef _Float16 v8h  __attribute__((ext_vector_type(8)));
typedef _Float16 v16h __attribute__((ext_vector_type(16)));
typedef _Float16 h4   __attribute__((ext_vector_type(4)));
typedef float    v8f  __attribute__((ext_vector_type(8)));

#define N_SEQ  512
#define EMB    64
#define HID    128
#define KDIM   128      // 2*EMB
#define M_TILE 128
#define LDK    136      // padded LDS row stride in f16 elements (272B, 16B aligned)

// ---------------------------------------------------------------------------
// Kernel 1: sL = s @ W_left + b_left ; sR = s @ W_right + b_right
// grid = B*N = 1024 blocks of 128 threads (tid<64 -> left, else right)
// ---------------------------------------------------------------------------
__global__ __launch_bounds__(128) void proj_kernel(
    const float* __restrict__ s,
    const float* __restrict__ Wl, const float* __restrict__ bl,
    const float* __restrict__ Wr, const float* __restrict__ br,
    float* __restrict__ outL, float* __restrict__ outR)
{
    const int bn  = blockIdx.x;            // 0..1023
    const int tid = threadIdx.x;
    const int d   = tid & 63;
    const bool left = (tid < 64);
    const float* __restrict__ W    = left ? Wl : Wr;
    const float* __restrict__ bias = left ? bl : br;
    float* __restrict__ o          = left ? outL : outR;

    const float* srow = s + (size_t)bn * 512;
    float acc = bias[d];
#pragma unroll 8
    for (int k = 0; k < 512; ++k)
        acc = fmaf(srow[k], W[k * 64 + d], acc);
    o[(size_t)bn * 64 + d] = acc;
}

// ---------------------------------------------------------------------------
// helpers
// ---------------------------------------------------------------------------
__device__ __forceinline__ v16h cat8(v8h lo, v8h hi) {
    return __builtin_shufflevector(lo, hi, 0,1,2,3,4,5,6,7,8,9,10,11,12,13,14,15);
}
__device__ __forceinline__ v8f splat8(float v) {
    v8f r;
#pragma unroll
    for (int i = 0; i < 8; ++i) r[i] = v;
    return r;
}
__device__ __forceinline__ h4 pack4(float a, float b, float c, float d) {
    h4 r; r[0] = (_Float16)a; r[1] = (_Float16)b; r[2] = (_Float16)c; r[3] = (_Float16)d;
    return r;
}

// ---------------------------------------------------------------------------
// Kernel 2: out[b][n][m][:] = relu(kcat @ W1 + b1) @ W2 + b2
// kcat[m][k] = k<64 ? sL[b,n,k]*sR[b,m,k] : emb[n-m+512][k-64]
// grid = B*N*(N/M_TILE) = 4096 blocks of 256 threads (8 waves)
// ---------------------------------------------------------------------------
__global__ __launch_bounds__(256) void pairwise_mlp_kernel(
    const float* __restrict__ sL, const float* __restrict__ sR,
    const float* __restrict__ emb,
    const float* __restrict__ W1, const float* __restrict__ b1,
    const float* __restrict__ W2, const float* __restrict__ b2,
    float* __restrict__ out)
{
    __shared__ __align__(16) _Float16 kcat[M_TILE * LDK];  // reused as h after GEMM1
    __shared__ __align__(16) _Float16 w1t [HID    * LDK];  // W1^T : [j][k]
    __shared__ __align__(16) _Float16 w2t [EMB    * LDK];  // W2^T : [j][k]

    const int tid = threadIdx.x;
    const int mt  = blockIdx.x & 3;
    const int n   = (blockIdx.x >> 2) & (N_SEQ - 1);
    const int b   = blockIdx.x >> 11;
    const int m0  = mt * M_TILE;

    // ---- stage kcat tile: 2 threads per row (half 0: product, half 1: emb)
    {
        const int row  = tid >> 1;          // 0..127
        const int half = tid & 1;
        const int m    = m0 + row;
        if (half == 0) {
            const float4* sLv = (const float4*)(sL + (size_t)(b * N_SEQ + n) * EMB);
            const float4* sRv = (const float4*)(sR + (size_t)(b * N_SEQ + m) * EMB);
            h4* dst = (h4*)&kcat[row * LDK];
#pragma unroll
            for (int i = 0; i < 16; ++i) {
                float4 a = sLv[i], c = sRv[i];
                dst[i] = pack4(a.x * c.x, a.y * c.y, a.z * c.z, a.w * c.w);
            }
        } else {
            const float4* ev = (const float4*)(emb + (size_t)(n - m + 512) * EMB);
            h4* dst = (h4*)&kcat[row * LDK + EMB];
#pragma unroll
            for (int i = 0; i < 16; ++i) {
                float4 e = ev[i];
                dst[i] = pack4(e.x, e.y, e.z, e.w);
            }
        }
    }
    // ---- stage W1^T (coalesced global read, strided LDS write)
#pragma unroll 4
    for (int i = 0; i < (HID * KDIM) / 256; ++i) {
        int idx = i * 256 + tid;
        int k = idx >> 7, j = idx & 127;
        w1t[j * LDK + k] = (_Float16)W1[idx];
    }
    // ---- stage W2^T
#pragma unroll 4
    for (int i = 0; i < (EMB * KDIM) / 256; ++i) {
        int idx = i * 256 + tid;
        int k = idx >> 6, j = idx & 63;
        w2t[j * LDK + k] = (_Float16)W2[idx];
    }
    __syncthreads();

    const int lane = tid & 31;
    const int wv   = tid >> 5;     // 0..7 : wave id
    const int l16  = lane & 15;
    const int hi   = lane >> 4;
    const int mloc = wv * 16;      // this wave's 16 private rows of kcat

    // ---- A fragments of GEMM1 (kept in VGPRs across all 8 HID tiles)
    // 16-bit A 16x32 layout: row = l16; lane-half hi selects K {0..7,16..23} vs {8..15,24..31}
    v16h a1[4];
#pragma unroll
    for (int kc = 0; kc < 4; ++kc) {
        const _Float16* base = &kcat[(mloc + l16) * LDK + kc * 32 + hi * 8];
        a1[kc] = cat8(*(const v8h*)base, *(const v8h*)(base + 16));
    }

    // ---- GEMM1: h = relu(kcat @ W1 + b1); write h f16 over wave-private kcat rows
#pragma unroll
    for (int t = 0; t < 8; ++t) {
        v8f acc = splat8(b1[t * 16 + l16]);   // C/D column N = l16 -> bias broadcast
#pragma unroll
        for (int kc = 0; kc < 4; ++kc) {
            // 16-bit B 32x16 layout: col = l16; lane-half hi selects K {0..7,16..23} vs {8..15,24..31}
            const _Float16* bb = &w1t[(t * 16 + l16) * LDK + kc * 32 + hi * 8];
            v16h bf = cat8(*(const v8h*)bb, *(const v8h*)(bb + 16));
            acc = __builtin_amdgcn_wmma_f32_16x16x32_f16(
                      false, a1[kc], false, bf, (short)0, acc, false, false);
        }
#pragma unroll
        for (int r = 0; r < 8; ++r) {         // C/D: M = hi*8 + r, N = l16
            float v = acc[r];
            v = v > 0.0f ? v : 0.0f;          // ReLU
            kcat[(mloc + hi * 8 + r) * LDK + t * 16 + l16] = (_Float16)v;
        }
    }

    // ---- A fragments of GEMM2 from h (same rows, now holding h)
    v16h a2[4];
#pragma unroll
    for (int kc = 0; kc < 4; ++kc) {
        const _Float16* base = &kcat[(mloc + l16) * LDK + kc * 32 + hi * 8];
        a2[kc] = cat8(*(const v8h*)base, *(const v8h*)(base + 16));
    }

    // ---- GEMM2: out = h @ W2 + b2
#pragma unroll
    for (int t = 0; t < 4; ++t) {
        v8f acc = splat8(b2[t * 16 + l16]);
#pragma unroll
        for (int kc = 0; kc < 4; ++kc) {
            const _Float16* bb = &w2t[(t * 16 + l16) * LDK + kc * 32 + hi * 8];
            v16h bf = cat8(*(const v8h*)bb, *(const v8h*)(bb + 16));
            acc = __builtin_amdgcn_wmma_f32_16x16x32_f16(
                      false, a2[kc], false, bf, (short)0, acc, false, false);
        }
#pragma unroll
        for (int r = 0; r < 8; ++r) {
            int m = m0 + mloc + hi * 8 + r;
            size_t off = (((size_t)(b * N_SEQ + n) * N_SEQ) + m) * EMB + t * 16 + l16;
            out[off] = acc[r];
        }
    }
}

// ---------------------------------------------------------------------------
extern "C" void kernel_launch(void* const* d_in, const int* in_sizes, int n_in,
                              void* d_out, int out_size, void* d_ws, size_t ws_size,
                              hipStream_t stream) {
    const float* s   = (const float*)d_in[0];
    const float* Wl  = (const float*)d_in[1];
    const float* bl  = (const float*)d_in[2];
    const float* Wr  = (const float*)d_in[3];
    const float* br  = (const float*)d_in[4];
    const float* emb = (const float*)d_in[5];
    const float* W1  = (const float*)d_in[6];
    const float* b1  = (const float*)d_in[7];
    const float* W2  = (const float*)d_in[8];
    const float* b2  = (const float*)d_in[9];
    float* out = (float*)d_out;

    float* sLw = (float*)d_ws;                 // [2,512,64] f32
    float* sRw = sLw + 2 * N_SEQ * EMB;        // [2,512,64] f32

    proj_kernel<<<2 * N_SEQ, 128, 0, stream>>>(s, Wl, bl, Wr, br, sLw, sRw);
    pairwise_mlp_kernel<<<2 * N_SEQ * (N_SEQ / M_TILE), 256, 0, stream>>>(
        sLw, sRw, emb, W1, b1, W2, b2, out);
}
